// VectorQuantizerGumbel_83648783057670
// MI455X (gfx1250) — matrix operand order repaired and checked by reading, not compiled, not run on hardware
//
#include <hip/hip_runtime.h>

// ---------------------------------------------------------------------------
// Gumbel-softmax Vector Quantizer, fused flash-style kernel for gfx1250.
//   z:        (16384, 256) f32   (B*C*P = 8*64*32 rows)
//   codebook: (8192, 256)  f32
// Outputs (concatenated in d_out, f32):
//   [0 .. 4194303]      z_q (straight-through value == z_q)
//   [4194304]           total_loss
//   [4194305 .. +16383] codes (argmax indices, as float)
//
// Numerics: logits ~ N(0,1) (z ~ N(0,1), codebook std 1/sqrt(256)); the hashed
// Gumbel noise is bounded to [-2.8, 17.3]. exp() with a fixed max of 0 is safe
// in f32/bf16, so no online-max rescaling is needed, and all cross-lane
// reductions (sums + argmax) are deferred to a single pass after the loop.
// ---------------------------------------------------------------------------

#define DIMN     256
#define NCODES   8192
#define NROWS    16384
#define CHUNK    32
#define ZQ_ELEMS (NROWS * DIMN)

typedef __attribute__((ext_vector_type(16))) __bf16 v16bf;
typedef __attribute__((ext_vector_type(8)))  float  v8f;

__device__ __forceinline__ unsigned short f2bf(float f) {
  union { float f; unsigned u; } v; v.f = f;
  unsigned r = v.u + 0x7fffu + ((v.u >> 16) & 1u);   // round-to-nearest-even
  return (unsigned short)(r >> 16);
}
__device__ __forceinline__ __bf16 us2bf(unsigned short u) {
  union { unsigned short u; __bf16 b; } x; x.u = u; return x.b;
}
__device__ __forceinline__ float hash_u01(unsigned x) {
  x ^= x >> 17; x *= 0xed5ad4bbu; x ^= x >> 11; x *= 0xac4c1b51u;
  x ^= x >> 15; x *= 0x31848babu; x ^= x >> 14;
  return ((float)(x >> 8) + 0.5f) * (1.0f / 16777216.0f);
}
__device__ __forceinline__ float gumbelf(unsigned idx) {
  float u = hash_u01(idx * 0x9E3779B9u + 0x85ebca6bu);
  return -__logf(-__logf(u));            // bounded to ~[-2.8, 17.3]
}

__global__ __launch_bounds__(128, 1)
void vq_main(const float* __restrict__ z, const float* __restrict__ cb,
             float* __restrict__ out, float* __restrict__ wacc) {
  // LDS staging: codebook chunk row-major (GEMM1 B) + transposed (GEMM2 B),
  // plus per-wave P-tile transpose buffers.
  __shared__ unsigned short cbS[CHUNK][DIMN + 8];    // [code][dim]
  __shared__ unsigned short cbT[DIMN][CHUNK + 8];    // [dim][code]
  __shared__ unsigned short pBuf[4][16][CHUNK + 8];  // [wave][row][code]

  const int tid  = threadIdx.x;
  const int wave = tid >> 5;
  const int lane = tid & 31;
  const int hf   = lane >> 4;     // lane half (C-layout rows 0-7 vs 8-15)
  const int ln   = lane & 15;
  const int row0 = blockIdx.x * 64 + wave * 16;

  // ---- preload A fragments: 16 rows x 256 dims of z as bf16 (ISA A layout)
  v16bf afrag[8];
  {
    const float* zr = z + (size_t)(row0 + ln) * DIMN + hf * 8;
#pragma unroll
    for (int kk = 0; kk < 8; ++kk) {
#pragma unroll
      for (int i = 0; i < 8; ++i) {
        int k = (i < 4) ? 2 * i : 16 + 2 * (i - 4);
        float2 v = *(const float2*)(zr + kk * 32 + k);
        afrag[kk][2 * i]     = us2bf(f2bf(v.x));
        afrag[kk][2 * i + 1] = us2bf(f2bf(v.y));
      }
    }
  }

  // ---- per-lane partial state (lane ln covers codes with c%16==ln)
  float s1[8], s2[8], sg[8], bestv[8];
  int   besti[8];
  v8f   accQ[16];
#pragma unroll
  for (int r = 0; r < 8; ++r) {
    s1[r] = 0.f; s2[r] = 0.f; sg[r] = 0.f;
    bestv[r] = -1e30f; besti[r] = 0;
  }
#pragma unroll
  for (int t = 0; t < 16; ++t) accQ[t] = (v8f){};

  for (int c0 = 0; c0 < NCODES; c0 += CHUNK) {
    // ---- stage codebook chunk into LDS (both layouts), f32 -> bf16
    {
      int r = tid >> 2;          // code row 0..31
      int q = tid & 3;           // dim quarter
      const float* src = cb + (size_t)(c0 + r) * DIMN + q * 64;
#pragma unroll
      for (int j = 0; j < 16; ++j) {
        float4 v = *(const float4*)(src + 4 * j);
        int d = q * 64 + 4 * j;
        unsigned short b0 = f2bf(v.x), b1 = f2bf(v.y), b2 = f2bf(v.z), b3 = f2bf(v.w);
        cbS[r][d] = b0; cbS[r][d + 1] = b1; cbS[r][d + 2] = b2; cbS[r][d + 3] = b3;
        cbT[d][r] = b0; cbT[d + 1][r] = b1; cbT[d + 2][r] = b2; cbT[d + 3][r] = b3;
      }
      if (c0 + CHUNK < NCODES)
        __builtin_prefetch(cb + (size_t)(c0 + CHUNK + r) * DIMN + q * 64, 0, 0);
    }
    __syncthreads();

    // ---- GEMM 1: logits tile L (16 rows x 32 codes), bf16 WMMA, f32 acc
    v8f L[2];
#pragma unroll
    for (int sub = 0; sub < 2; ++sub) {
      v8f Lt = (v8f){};
#pragma unroll
      for (int kk = 0; kk < 8; ++kk) {
        v16bf b;
        const unsigned short* bp = &cbS[sub * 16 + ln][kk * 32 + hf * 8];
#pragma unroll
        for (int i = 0; i < 8; ++i) {
          int k = (i < 4) ? 2 * i : 16 + 2 * (i - 4);
          b[2 * i]     = us2bf(bp[k]);
          b[2 * i + 1] = us2bf(bp[k + 1]);
        }
        Lt = __builtin_amdgcn_wmma_f32_16x16x32_bf16(
            false, afrag[kk], false, b, (short)0, Lt, false, false);
      }
      L[sub] = Lt;
    }

    // ---- per-element softmax terms; no cross-lane work in the hot loop
#pragma unroll
    for (int r = 0; r < 8; ++r) {
      const int grow  = row0 + hf * 8 + r;
      const int code0 = c0 + ln, code1 = c0 + 16 + ln;
      float l0 = L[0][r], l1 = L[1][r];
      float lg0 = l0 + gumbelf((unsigned)grow * (unsigned)NCODES + (unsigned)code0);
      float lg1 = l1 + gumbelf((unsigned)grow * (unsigned)NCODES + (unsigned)code1);

      float e0 = __expf(l0), e1 = __expf(l1);
      s1[r] += e0 + e1;
      s2[r] += l0 * e0 + l1 * e1;

      float w0 = __expf(lg0), w1 = __expf(lg1);   // TAU = 1
      sg[r] += w0 + w1;

      if (l0 > bestv[r]) { bestv[r] = l0; besti[r] = code0; }
      if (l1 > bestv[r]) { bestv[r] = l1; besti[r] = code1; }

      // unnormalized Gumbel weights -> P buffer (C-layout row, both subtiles)
      pBuf[wave][hf * 8 + r][ln]      = f2bf(w0);
      pBuf[wave][hf * 8 + r][16 + ln] = f2bf(w1);
    }
    asm volatile("" ::: "memory");  // keep LDS write->read order

    // ---- GEMM 2: accQ(16x256) += P(16x32) @ codebook_chunk(32x256)
    v16bf a2;
#pragma unroll
    for (int i = 0; i < 8; ++i) {
      int k = ((i < 4) ? 2 * i : 16 + 2 * (i - 4)) + 8 * hf;
      a2[2 * i]     = us2bf(pBuf[wave][ln][k]);
      a2[2 * i + 1] = us2bf(pBuf[wave][ln][k + 1]);
    }
#pragma unroll
    for (int t = 0; t < 16; ++t) {
      v16bf b2;
      const unsigned short* bp = &cbT[t * 16 + ln][hf * 8];
#pragma unroll
      for (int i = 0; i < 8; ++i) {
        int k = (i < 4) ? 2 * i : 16 + 2 * (i - 4);
        b2[2 * i]     = us2bf(bp[k]);
        b2[2 * i + 1] = us2bf(bp[k + 1]);
      }
      accQ[t] = __builtin_amdgcn_wmma_f32_16x16x32_bf16(
          false, a2, false, b2, (short)0, accQ[t], false, false);
    }
    __syncthreads();  // before next chunk overwrites cbS/cbT
  }

  // ---- one-time cross-lane reductions over the 16 lanes of each row group
#pragma unroll
  for (int r = 0; r < 8; ++r) {
#pragma unroll
    for (int off = 8; off >= 1; off >>= 1) {
      s1[r] += __shfl_xor(s1[r], off, 32);
      s2[r] += __shfl_xor(s2[r], off, 32);
      sg[r] += __shfl_xor(sg[r], off, 32);
      float ov = __shfl_xor(bestv[r], off, 32);
      int   oi = __shfl_xor(besti[r], off, 32);
      if (ov > bestv[r] || (ov == bestv[r] && oi < besti[r])) {
        bestv[r] = ov; besti[r] = oi;
      }
    }
  }

  // ---- finalize: z_q = accQ / sg, commit-loss partial sum, codes, KL
  float locsq = 0.f;
#pragma unroll
  for (int t = 0; t < 16; ++t) {
#pragma unroll
    for (int r = 0; r < 8; ++r) {
      const int grow = row0 + hf * 8 + r;
      const int d    = t * 16 + ln;
      size_t idx = (size_t)grow * DIMN + d;
      float q = accQ[t][r] / sg[r];
      float diff = q - z[idx];
      locsq += diff * diff;
      out[idx] = q;
    }
  }
#pragma unroll
  for (int off = 16; off >= 1; off >>= 1) locsq += __shfl_xor(locsq, off, 32);
  if (lane == 0) atomicAdd(&wacc[0], locsq);

  if (ln == 0) {
    float lk = 0.f;
#pragma unroll
    for (int r = 0; r < 8; ++r) {
      // max fixed at 0: lse = log(s1); KL row = E[l] - lse + log K
      lk += s2[r] / s1[r] - __logf(s1[r]) + __logf((float)NCODES);
      const int grow = row0 + hf * 8 + r;
      out[(size_t)ZQ_ELEMS + 1 + grow] = (float)besti[r];
    }
    atomicAdd(&wacc[1], lk);
  }
}

__global__ void vq_init(float* wacc) {
  if (threadIdx.x < 2) wacc[threadIdx.x] = 0.f;
}

__global__ void vq_fin(const float* wacc, float* out) {
  if (threadIdx.x == 0)
    out[ZQ_ELEMS] = 0.25f * wacc[0] / (float)ZQ_ELEMS +
                    0.01f * wacc[1] / (float)NROWS;
}

extern "C" void kernel_launch(void* const* d_in, const int* in_sizes, int n_in,
                              void* d_out, int out_size, void* d_ws, size_t ws_size,
                              hipStream_t stream) {
  const float* z  = (const float*)d_in[0];
  const float* cb = (const float*)d_in[1];
  float* out  = (float*)d_out;
  float* wacc = (float*)d_ws;
  hipLaunchKernelGGL(vq_init, dim3(1), dim3(32), 0, stream, wacc);
  hipLaunchKernelGGL(vq_main, dim3(NROWS / 64), dim3(128), 0, stream, z, cb, out, wacc);
  hipLaunchKernelGGL(vq_fin, dim3(1), dim3(32), 0, stream, wacc, out);
}
